// MultiScaleDynamicConv_25013889532389
// MI455X (gfx1250) — matrix operand order repaired and checked by reading, not compiled, not run on hardware
//
#include <hip/hip_runtime.h>
#include <hip/hip_bf16.h>

// ---------------------------------------------------------------------------
// Problem constants (from reference): B=32, C=256, H=W=64, Cr=64
// ---------------------------------------------------------------------------
#define BB   32
#define CC   256
#define HH   64
#define WW   64
#define HW   (HH * WW)          // 4096
#define PAD  3
#define TW   (WW + 2 * PAD)     // 70
#define TH   (HH + 2 * PAD)     // 70
#define TS   72                 // padded LDS row stride (bank-friendly)
#define K3C  (3 * CC)           // 768

typedef __attribute__((ext_vector_type(16))) __bf16 v16bf;
typedef __attribute__((ext_vector_type(8)))  float  v8f;

// ---------------------------------------------------------------------------
// Kernel 1: fused depthwise 3x3 / 5x5 / 7x7 conv + per-channel spatial means.
// One block per (b, c) image. 256 threads. 70x70 halo tile in LDS.
// Each thread computes 16 consecutive pixels of one row using a sliding
// 22-wide register row buffer (reuses LDS reads across the 7 taps in x).
// ---------------------------------------------------------------------------
__global__ __launch_bounds__(256) void dwconv_kernel(
    const float* __restrict__ x,
    const float* __restrict__ w3, const float* __restrict__ b3,
    const float* __restrict__ w5, const float* __restrict__ b5,
    const float* __restrict__ w7, const float* __restrict__ b7,
    float* __restrict__ x3o, float* __restrict__ x5o, float* __restrict__ x7o,
    float* __restrict__ m)   // (B, 3C) channel means of x_cat
{
    __shared__ float tile[TH * TS];
    __shared__ float w3s[9], w5s[25], w7s[49];
    __shared__ float red[256];

    const int tid = threadIdx.x;
    const int bc  = blockIdx.x;          // b*C + c
    const int b   = bc >> 8;
    const int c   = bc & (CC - 1);
    const int imgbase = bc * HW;

    // --- stage weights for this channel into LDS (uniform per block) ---
    if (tid < 9)  w3s[tid] = w3[c * 9  + tid];
    if (tid < 25) w5s[tid] = w5[c * 25 + tid];
    if (tid < 49) w7s[tid] = w7[c * 49 + tid];

    // --- load 70x70 halo tile (zero padded) ---
    for (int i = tid; i < TH * TW; i += 256) {
        const int ty = i / TW;
        const int tx = i - ty * TW;
        const int gy = ty - PAD;
        const int gx = tx - PAD;
        float v = 0.0f;
        if (gy >= 0 && gy < HH && gx >= 0 && gx < WW)
            v = x[imgbase + gy * WW + gx];
        tile[ty * TS + tx] = v;
    }
    __syncthreads();

    const float bias3 = b3[c], bias5 = b5[c], bias7 = b7[c];

    // thread -> (row, 16-wide x segment)
    const int row = tid >> 2;            // 0..63
    const int x0  = (tid & 3) * 16;      // 0,16,32,48

    float s3[16], s5[16], s7[16];
    #pragma unroll
    for (int i = 0; i < 16; ++i) { s3[i] = bias3; s5[i] = bias5; s7[i] = bias7; }

    #pragma unroll
    for (int dy = -3; dy <= 3; ++dy) {
        // sliding register row buffer: tile cols [x0 .. x0+21] of row (row+3+dy)
        float rb[22];
        const int rbase = (row + PAD + dy) * TS + x0;
        #pragma unroll
        for (int j = 0; j < 22; ++j) rb[j] = tile[rbase + j];

        #pragma unroll
        for (int dx = -3; dx <= 3; ++dx) {
            const float c7 = w7s[(dy + 3) * 7 + (dx + 3)];
            #pragma unroll
            for (int i = 0; i < 16; ++i) s7[i] = fmaf(rb[i + dx + 3], c7, s7[i]);
            if (dy >= -2 && dy <= 2 && dx >= -2 && dx <= 2) {
                const float c5 = w5s[(dy + 2) * 5 + (dx + 2)];
                #pragma unroll
                for (int i = 0; i < 16; ++i) s5[i] = fmaf(rb[i + dx + 3], c5, s5[i]);
            }
            if (dy >= -1 && dy <= 1 && dx >= -1 && dx <= 1) {
                const float c3 = w3s[(dy + 1) * 3 + (dx + 1)];
                #pragma unroll
                for (int i = 0; i < 16; ++i) s3[i] = fmaf(rb[i + dx + 3], c3, s3[i]);
            }
        }
    }

    // --- vectorized stores (16 consecutive, 16-element aligned) ---
    const int obase = imgbase + row * WW + x0;
    #pragma unroll
    for (int q = 0; q < 4; ++q) {
        ((float4*)(x3o + obase))[q] = make_float4(s3[4*q], s3[4*q+1], s3[4*q+2], s3[4*q+3]);
        ((float4*)(x5o + obase))[q] = make_float4(s5[4*q], s5[4*q+1], s5[4*q+2], s5[4*q+3]);
        ((float4*)(x7o + obase))[q] = make_float4(s7[4*q], s7[4*q+1], s7[4*q+2], s7[4*q+3]);
    }

    // --- block reductions for channel means (x_cat order: [x3 | x5 | x7]) ---
    float psum[3] = {0.f, 0.f, 0.f};
    #pragma unroll
    for (int i = 0; i < 16; ++i) { psum[0] += s3[i]; psum[1] += s5[i]; psum[2] += s7[i]; }

    for (int s = 0; s < 3; ++s) {
        __syncthreads();
        red[tid] = psum[s];
        __syncthreads();
        for (int off = 128; off > 0; off >>= 1) {
            if (tid < off) red[tid] += red[tid + off];
            __syncthreads();
        }
        if (tid == 0)
            m[b * K3C + s * CC + c] = red[0] * (1.0f / (float)HW);
    }
}

// ---------------------------------------------------------------------------
// Kernel 2: g = m @ w_pw^T + b_pw   via v_wmma_f32_16x16x32_bf16
//   A = m (32 x 768), B[k,n] = w_pw[n,k] (768 x 256), D = g (32 x 256)
// 32 tiles of 16x16 (2 M-tiles x 16 N-tiles); one wave per tile; K in 32-steps.
// Fragment packing follows the CDNA5 ISA 16-bit A (16x32) and B (32x16)
// wave32 VGPR layouts.
// ---------------------------------------------------------------------------
__global__ __launch_bounds__(32) void gemm_g_kernel(
    const float* __restrict__ m,
    const float* __restrict__ w_pw,
    const float* __restrict__ b_pw,
    float* __restrict__ g)
{
    const int tileId = blockIdx.x;       // 0..31
    const int mt = tileId >> 4;          // 0..1
    const int nt = tileId & 15;          // 0..15
    const int lane = threadIdx.x;        // 0..31 (one wave)
    const int r  = lane & 15;
    const int hh = lane >> 4;

    v8f acc = {};

    #pragma unroll 4
    for (int kb = 0; kb < K3C; kb += 32) {
        v16bf afrag, bfrag;
        #pragma unroll
        for (int e = 0; e < 16; ++e) {
            // A 16x32 bf16: lane r = row M; VGPR v holds K-pairs:
            //   v<4 : K = 2v + 8*hh ; v>=4 : K = 16 + 2(v-4) + 8*hh
            const int v  = e >> 1;
            const int ka = ((v < 4) ? (2 * v + 8 * hh) : (16 + 2 * (v - 4) + 8 * hh)) + (e & 1);
            afrag[e] = (__bf16)m[(mt * 16 + r) * K3C + kb + ka];
            // B 32x16 bf16: lane r = col N; element e -> K = e + 8*hh (+8 if e>=8)
            const int kb2 = e + 8 * hh + ((e >= 8) ? 8 : 0);
            bfrag[e] = (__bf16)w_pw[(nt * 16 + r) * K3C + kb + kb2];
        }
        acc = __builtin_amdgcn_wmma_f32_16x16x32_bf16(
            /*neg_a=*/false, afrag, /*neg_b=*/false, bfrag,
            /*c_mod=*/(short)0, acc, /*reuse_a=*/false, /*reuse_b=*/false);
    }

    // D layout: VGPR rr, lane -> (M = rr + 8*hh, N = lane&15)
    #pragma unroll
    for (int rr = 0; rr < 8; ++rr) {
        const int gb = mt * 16 + rr + 8 * hh;   // batch row
        const int go = nt * 16 + r;             // output channel
        g[gb * CC + go] = acc[rr] + b_pw[go];
    }
}

// ---------------------------------------------------------------------------
// Kernel 3: SE attention head: h = mish(g @ w_a1^T + b_a1); logits; softmax.
// One 64-thread block per batch element.
// ---------------------------------------------------------------------------
__global__ __launch_bounds__(64) void attn_kernel(
    const float* __restrict__ g,
    const float* __restrict__ w_a1, const float* __restrict__ b_a1,
    const float* __restrict__ w_a2, const float* __restrict__ b_a2,
    float* __restrict__ wts)
{
    __shared__ float hs[64];
    __shared__ float lg[3];
    const int b = blockIdx.x;
    const int j = threadIdx.x;

    float acc = b_a1[j];
    for (int c = 0; c < CC; ++c)
        acc = fmaf(g[b * CC + c], w_a1[j * CC + c], acc);
    // mish(x) = x * tanh(softplus(x))
    const float sp = (acc > 20.0f) ? acc : log1pf(__expf(acc));
    hs[j] = acc * tanhf(sp);
    __syncthreads();

    if (j < 3) {
        float l = b_a2[j];
        for (int t = 0; t < 64; ++t)
            l = fmaf(hs[t], w_a2[j * 64 + t], l);
        lg[j] = l;
    }
    __syncthreads();

    if (j == 0) {
        const float mx = fmaxf(lg[0], fmaxf(lg[1], lg[2]));
        const float e0 = __expf(lg[0] - mx);
        const float e1 = __expf(lg[1] - mx);
        const float e2 = __expf(lg[2] - mx);
        const float inv = 1.0f / (e0 + e1 + e2);
        wts[b * 3 + 0] = e0 * inv;
        wts[b * 3 + 1] = e1 * inv;
        wts[b * 3 + 2] = e2 * inv;
    }
}

// ---------------------------------------------------------------------------
// Kernel 4: out = w0*x3 + w1*x5 + w2*x7  (float4 streaming, HBM-bound)
// ---------------------------------------------------------------------------
__global__ __launch_bounds__(256) void combine_kernel(
    const float* __restrict__ x3, const float* __restrict__ x5,
    const float* __restrict__ x7, const float* __restrict__ wts,
    float* __restrict__ out)
{
    const int idx = blockIdx.x * 256 + threadIdx.x;        // float4 index
    const int n4 = (BB * CC * HW) / 4;                     // 8388608
    if (idx >= n4) return;
    const int b = (idx * 4) / (CC * HW);
    const float w0 = wts[b * 3 + 0];
    const float w1 = wts[b * 3 + 1];
    const float w2 = wts[b * 3 + 2];
    const float4 a = ((const float4*)x3)[idx];
    const float4 c = ((const float4*)x5)[idx];
    const float4 d = ((const float4*)x7)[idx];
    float4 o;
    o.x = fmaf(w0, a.x, fmaf(w1, c.x, w2 * d.x));
    o.y = fmaf(w0, a.y, fmaf(w1, c.y, w2 * d.y));
    o.z = fmaf(w0, a.z, fmaf(w1, c.z, w2 * d.z));
    o.w = fmaf(w0, a.w, fmaf(w1, c.w, w2 * d.w));
    ((float4*)out)[idx] = o;
}

// ---------------------------------------------------------------------------
// Host launcher
// ---------------------------------------------------------------------------
extern "C" void kernel_launch(void* const* d_in, const int* in_sizes, int n_in,
                              void* d_out, int out_size, void* d_ws, size_t ws_size,
                              hipStream_t stream) {
    const float* x    = (const float*)d_in[0];
    const float* w3   = (const float*)d_in[1];
    const float* b3   = (const float*)d_in[2];
    const float* w5   = (const float*)d_in[3];
    const float* b5   = (const float*)d_in[4];
    const float* w7   = (const float*)d_in[5];
    const float* b7   = (const float*)d_in[6];
    const float* w_pw = (const float*)d_in[7];
    const float* b_pw = (const float*)d_in[8];
    const float* w_a1 = (const float*)d_in[9];
    const float* b_a1 = (const float*)d_in[10];
    const float* w_a2 = (const float*)d_in[11];
    const float* b_a2 = (const float*)d_in[12];
    float* out = (float*)d_out;

    float* ws = (float*)d_ws;
    const size_t N = (size_t)BB * CC * HW;   // 33,554,432
    float* x3  = ws;
    float* x5  = ws + N;
    float* x7  = ws + 2 * N;
    float* m   = ws + 3 * N;                 // (32, 768)
    float* g   = m + BB * K3C;               // (32, 256)
    float* wts = g + BB * CC;                // (32, 3)

    dwconv_kernel<<<BB * CC, 256, 0, stream>>>(x, w3, b3, w5, b5, w7, b7,
                                               x3, x5, x7, m);
    gemm_g_kernel<<<32, 32, 0, stream>>>(m, w_pw, b_pw, g);
    attn_kernel<<<BB, 64, 0, stream>>>(g, w_a1, b_a1, w_a2, b_a2, wts);
    combine_kernel<<<(int)((N / 4 + 255) / 256), 256, 0, stream>>>(x3, x5, x7, wts, out);
}
